// PrototypeGraphNCD_5016521802334
// MI455X (gfx1250) — compile-verified, matrix operand order checked by reading, not run on hardware
//
#include <hip/hip_runtime.h>
#include <math.h>

typedef unsigned short u16;
typedef unsigned int   u32;
typedef __attribute__((ext_vector_type(16))) __bf16          v16bf;
typedef __attribute__((ext_vector_type(16))) unsigned short  v16us;
typedef __attribute__((ext_vector_type(8)))  float           v8f;
typedef __attribute__((ext_vector_type(4)))  unsigned int    u32x4;
typedef __attribute__((ext_vector_type(8)))  int             i32x8;
typedef __attribute__((ext_vector_type(4)))  int             i32x4;

#define NPTS 131072
#define DDIM 512
#define HDIM 512
#define NKP  20
#define NVP  10
#define NP   30
#define EPSV 1e-8f

__device__ __forceinline__ u16 f2bf(float f) {
  u32 u = __float_as_uint(f);
  u32 r = u + 0x7FFFu + ((u >> 16) & 1u);
  return (u16)(r >> 16);
}
__device__ __forceinline__ float bf2f(u16 h) {
  return __uint_as_float(((u32)h) << 16);
}
__device__ __forceinline__ float warpSum(float v) {
#pragma unroll
  for (int o = 16; o > 0; o >>= 1) v += __shfl_xor(v, o, 32);
  return v;
}

// ---- Tensor Data Mover: 2D bf16 tile load with LDS row padding ----
// D# packing per CDNA5 ISA.
__device__ __forceinline__ void tdm_load_tile_bf16(
    unsigned ldsAddr, const u16* gptr,
    unsigned tensorD0, unsigned tensorD1, unsigned long long strideD0,
    unsigned tileD0, unsigned tileD1,
    int padIntervalCode, int padAmountCode, int padEnable) {
  unsigned long long ga = (unsigned long long)(size_t)gptr;
  u32x4 g0;
  g0[0] = 1u;                                   // count=1 (valid), user mode
  g0[1] = ldsAddr;                              // LDS byte address
  g0[2] = (unsigned)ga;                         // global_addr[31:0]
  g0[3] = (unsigned)((ga >> 32) & 0x01FFFFFFull) | (2u << 30);  // addr[56:32] | type=2
  i32x8 g1;
  g1[0] = (int)((1u << 16)                      // data_size = 1 -> 2 bytes
                | (padEnable ? (1u << 20) : 0u)
                | ((unsigned)padIntervalCode << 22)
                | ((unsigned)padAmountCode << 25));
  g1[1] = (int)((tensorD0 & 0xFFFFu) << 16);    // tensor_dim0[15:0] (bits 63:48)
  g1[2] = (int)(((tensorD0 >> 16) & 0xFFFFu) | ((tensorD1 & 0xFFFFu) << 16));
  g1[3] = (int)(((tensorD1 >> 16) & 0xFFFFu) | ((tileD0 & 0xFFFFu) << 16));
  g1[4] = (int)(tileD1 & 0xFFFFu);              // tile_dim1, tile_dim2=0
  g1[5] = (int)(unsigned)(strideD0 & 0xFFFFFFFFull);
  g1[6] = (int)(unsigned)((strideD0 >> 32) & 0xFFFFull);
  g1[7] = 0;
  i32x4 gz = {0, 0, 0, 0};
#if defined(__clang_major__) && (__clang_major__ >= 23)
  i32x8 gz8 = {0, 0, 0, 0, 0, 0, 0, 0};
  __builtin_amdgcn_tensor_load_to_lds(g0, g1, gz, gz, gz8, 0);
#else
  __builtin_amdgcn_tensor_load_to_lds(g0, g1, gz, gz, 0);
#endif
}

// ---------------------------------------------------------------- utilities
__global__ void k_zero(float* p, int n) {
  int i = blockIdx.x * blockDim.x + threadIdx.x;
  if (i < n) p[i] = 0.f;
}

// W (K x N, f32) -> Wt (N x K, bf16), LDS-tiled transpose
__global__ void k_transpose_w(const float* __restrict__ src, u16* __restrict__ dst) {
  __shared__ float tile[32][33];
  const int bk = blockIdx.x * 32;   // k base
  const int bn = blockIdx.y * 32;   // n base
  const int tx = threadIdx.x & 31, ty = threadIdx.x >> 5;  // 32 x 8
  for (int i = ty; i < 32; i += 8)
    tile[i][tx] = src[(size_t)(bk + i) * DDIM + bn + tx];  // [k_local][n_local]
  __syncthreads();
  for (int i = ty; i < 32; i += 8)
    dst[(size_t)(bn + i) * DDIM + bk + tx] = f2bf(tile[tx][i]);
}

// normalize the 30 prototypes (rows of length 512)
__global__ void k_norm_protos(const float* __restrict__ kp, const float* __restrict__ nv,
                              float* __restrict__ protoN) {
  int p = blockIdx.x, tid = threadIdx.x;
  const float* src = (p < NKP) ? kp + (size_t)p * DDIM : nv + (size_t)(p - NKP) * DDIM;
  __shared__ float red[256];
  float s = 0.f;
  for (int i = tid; i < DDIM; i += 256) { float v = src[i]; s += v * v; }
  red[tid] = s; __syncthreads();
  for (int o = 128; o > 0; o >>= 1) { if (tid < o) red[tid] += red[tid + o]; __syncthreads(); }
  float inv = 1.0f / fmaxf(sqrtf(red[0]), EPSV);
  for (int i = tid; i < DDIM; i += 256) protoN[(size_t)p * DDIM + i] = src[i] * inv;
}

// ------------------------------------------------------------------- phase1
__global__ void k_phase1(const float* __restrict__ pf, const float* __restrict__ kp,
                         const float* __restrict__ nv, const float* __restrict__ protoN,
                         const float* __restrict__ tempPtr,
                         float* __restrict__ outAdj, float* __restrict__ colsum,
                         u16* __restrict__ a_in, u16* __restrict__ contrib) {
  __shared__ float sPN[NP * DDIM];
  __shared__ float sCS[NP];
  const int tid = threadIdx.x;
  for (int i = tid; i < NP * DDIM; i += 256) sPN[i] = protoN[i];
  if (tid < NP) sCS[tid] = 0.f;
  __syncthreads();
  const int lane = tid & 31, wid = tid >> 5;
  const float invT = 1.0f / tempPtr[0];

  for (int rr = 0; rr < 16; ++rr) {
    const size_t r = (size_t)blockIdx.x * 128 + wid * 16 + rr;
    float x[16];
    const float4* prow = reinterpret_cast<const float4*>(pf + r * DDIM + lane * 16);
#pragma unroll
    for (int q = 0; q < 4; ++q) {
      float4 f = prow[q];
      x[4 * q] = f.x; x[4 * q + 1] = f.y; x[4 * q + 2] = f.z; x[4 * q + 3] = f.w;
    }
    float ss = 0.f;
#pragma unroll
    for (int j = 0; j < 16; ++j) ss += x[j] * x[j];
    ss = warpSum(ss);
    const float inv = 1.0f / fmaxf(sqrtf(ss), EPSV);

    float sims[NP];
#pragma unroll
    for (int p = 0; p < NP; ++p) {
      const float* pr = &sPN[p * DDIM + lane * 16];
      float pa = 0.f;
#pragma unroll
      for (int j = 0; j < 16; ++j) pa += x[j] * pr[j];
      sims[p] = warpSum(pa) * inv * invT;
    }

    float vals[5]; int idx[5]; u32 chosen = 0;
#pragma unroll
    for (int t = 0; t < 5; ++t) {
      float bv = -INFINITY; int bi = 0;
#pragma unroll
      for (int p = 0; p < NP; ++p)
        if (!((chosen >> p) & 1u) && sims[p] > bv) { bv = sims[p]; bi = p; }
      vals[t] = bv; idx[t] = bi; chosen |= (1u << bi);
    }
    float s = 0.f;
#pragma unroll
    for (int t = 0; t < 5; ++t) s += vals[t];
    const float den = (s > 0.f) ? s : 1.0f;

    if (lane < NP) {
      float wl = 0.f;
#pragma unroll
      for (int t = 0; t < 5; ++t) if (idx[t] == lane) wl = vals[t] / den;
      outAdj[r * NP + lane] = wl;
      atomicAdd(&sCS[lane], wl);
    }

    float c[16];
#pragma unroll
    for (int j = 0; j < 16; ++j) c[j] = 0.f;
#pragma unroll
    for (int t = 0; t < 5; ++t) {
      int p = idx[t]; float w = vals[t] / den;
      const float* pr = (p < NKP) ? (kp + (size_t)p * DDIM) : (nv + (size_t)(p - NKP) * DDIM);
      const float4* p4 = reinterpret_cast<const float4*>(pr + lane * 16);
#pragma unroll
      for (int q = 0; q < 4; ++q) {
        float4 f = p4[q];
        c[4 * q] += w * f.x; c[4 * q + 1] += w * f.y;
        c[4 * q + 2] += w * f.z; c[4 * q + 3] += w * f.w;
      }
    }

    u16 ab[16] __attribute__((aligned(16)));
    u16 cb[16] __attribute__((aligned(16)));
#pragma unroll
    for (int j = 0; j < 16; ++j) { ab[j] = f2bf(x[j] + c[j]); cb[j] = f2bf(c[j]); }
    uint4* da = reinterpret_cast<uint4*>(a_in + r * DDIM + lane * 16);
    uint4* dc = reinterpret_cast<uint4*>(contrib + r * DDIM + lane * 16);
    da[0] = ((uint4*)ab)[0]; da[1] = ((uint4*)ab)[1];
    dc[0] = ((uint4*)cb)[0]; dc[1] = ((uint4*)cb)[1];
  }
  __syncthreads();
  if (tid < NP) atomicAdd(&colsum[tid], sCS[tid]);
}

// ------------------------------------------------------------- WMMA GEMM
// C = act(A(bf16) @ Wt^T(bf16) + bias) [+ extra], K=512; Wt is N x K.
// 128x64 tile, BK=32; BOTH tiles staged by TDM (double-buffered, padded to
// stride-40 rows); one barrier per K step; 8 waves x 4 wmma per step.
#define LOAD_FRAG(dst, ARR, ROW0)                                            \
  do {                                                                       \
    v16us _u; int _r = (ROW0) + (lane & 15); int _kb = ((lane >> 4) << 3);   \
    _Pragma("unroll")                                                        \
    for (int _v = 0; _v < 8; ++_v) {                                         \
      int _k = ((_v < 4) ? 0 : 16) + _kb + ((_v & 3) << 1);                  \
      u32 _pr = *reinterpret_cast<const u32*>(&ARR[_r * 40 + _k]);           \
      _u[2 * _v] = (u16)(_pr & 0xFFFFu); _u[2 * _v + 1] = (u16)(_pr >> 16);  \
    }                                                                        \
    dst = __builtin_bit_cast(v16bf, _u);                                     \
  } while (0)

template <bool RELU, bool HAS_EXTRA, bool OUT_BF, bool OUT_F>
__global__ void k_gemm(const u16* __restrict__ A, const u16* __restrict__ Wt,
                       const float* __restrict__ bias, const u16* __restrict__ extra,
                       u16* __restrict__ outBf, float* __restrict__ outF) {
  __shared__ __align__(16) u16 As[2][128 * 40];  // M x K, TDM-padded rows
  __shared__ __align__(16) u16 Bs[2][64 * 40];   // N x K, TDM-padded rows
  const int tid = threadIdx.x;
  const int lane = tid & 31;
  const int wid = tid >> 5;
  const int m0 = (wid & 3) * 32;
  const int n0w = (wid >> 2) * 32;
  const size_t rowBlk = (size_t)blockIdx.x * 128;
  const int colBlk = blockIdx.y * 64;

  auto issueTiles = [&](int kt, int buf) {
    // A: 128 rows x 32 bf16, row stride 512 -> [128][40]
    tdm_load_tile_bf16((unsigned)(size_t)(void*)&As[buf][0],
                       A + rowBlk * DDIM + (size_t)kt * 32,
                       DDIM, NPTS, DDIM, 32, 128, /*padInt=*/3, /*padAmt=*/3, 1);
    // Wt: 64 rows x 32 bf16, row stride 512 -> [64][40]
    tdm_load_tile_bf16((unsigned)(size_t)(void*)&Bs[buf][0],
                       Wt + (size_t)colBlk * DDIM + (size_t)kt * 32,
                       DDIM, HDIM, DDIM, 32, 64, /*padInt=*/3, /*padAmt=*/3, 1);
  };

  if (wid == 0) issueTiles(0, 0);

  v8f acc00 = {0,0,0,0,0,0,0,0}, acc01 = {0,0,0,0,0,0,0,0};
  v8f acc10 = {0,0,0,0,0,0,0,0}, acc11 = {0,0,0,0,0,0,0,0};

  for (int kt = 0; kt < 16; ++kt) {
    const int cur = kt & 1, nxt = cur ^ 1;
    if (wid == 0) __builtin_amdgcn_s_wait_tensorcnt(0);  // tiles[cur] in LDS
    __syncthreads();  // also proves all waves finished reading tiles[nxt]
    if (wid == 0 && kt < 15) issueTiles(kt + 1, nxt);    // overlaps compute

    v16bf a0, a1, b0, b1;
    LOAD_FRAG(a0, As[cur], m0);
    LOAD_FRAG(a1, As[cur], m0 + 16);
    LOAD_FRAG(b0, Bs[cur], n0w);
    LOAD_FRAG(b1, Bs[cur], n0w + 16);
    acc00 = __builtin_amdgcn_wmma_f32_16x16x32_bf16(false, a0, false, b0, (short)0, acc00, false, false);
    acc01 = __builtin_amdgcn_wmma_f32_16x16x32_bf16(false, a0, false, b1, (short)0, acc01, false, false);
    acc10 = __builtin_amdgcn_wmma_f32_16x16x32_bf16(false, a1, false, b0, (short)0, acc10, false, false);
    acc11 = __builtin_amdgcn_wmma_f32_16x16x32_bf16(false, a1, false, b1, (short)0, acc11, false, false);
  }

#define EPILOG(ACC, MS, NS)                                                  \
  do {                                                                       \
    int cc = colBlk + n0w + (NS) + (lane & 15);                              \
    float bv = bias[cc];                                                     \
    int rbase = (int)rowBlk + m0 + (MS) + ((lane >> 4) << 3);                \
    _Pragma("unroll")                                                        \
    for (int _v = 0; _v < 8; ++_v) {                                         \
      int rrow = rbase + _v;                                                 \
      float val = ACC[_v] + bv;                                              \
      if (RELU) val = fmaxf(val, 0.0f);                                      \
      size_t gi = (size_t)rrow * HDIM + cc;                                  \
      if (HAS_EXTRA) val += bf2f(extra[gi]);                                 \
      if (OUT_BF) outBf[gi] = f2bf(val);                                     \
      if (OUT_F)  outF[gi] = val;                                            \
    }                                                                        \
  } while (0)

  EPILOG(acc00, 0, 0);
  EPILOG(acc01, 0, 16);
  EPILOG(acc10, 16, 0);
  EPILOG(acc11, 16, 16);
#undef EPILOG
}

// ------------------------------------------------ adj^T @ refined (30x512)
__global__ void k_proto_proj(const float* __restrict__ adj, const float* __restrict__ refined,
                             float* __restrict__ kwnw) {
  const int tid = threadIdx.x;
  const int d0 = tid * 2;
  float a0[NP], a1[NP];
#pragma unroll
  for (int p = 0; p < NP; ++p) { a0[p] = 0.f; a1[p] = 0.f; }
  __shared__ float adjBuf[8 * NP];
  const size_t base = (size_t)blockIdx.x * 1024;
  for (int g = 0; g < 1024; g += 8) {
    __syncthreads();
    if (tid < 8 * NP) adjBuf[tid] = adj[(base + g + tid / NP) * NP + (tid % NP)];
    __syncthreads();
    for (int rr = 0; rr < 8; ++rr) {
      const float2 rv = *reinterpret_cast<const float2*>(refined + (base + g + rr) * DDIM + d0);
#pragma unroll
      for (int p = 0; p < NP; ++p) {
        float wv = adjBuf[rr * NP + p];
        a0[p] += wv * rv.x; a1[p] += wv * rv.y;
      }
    }
  }
#pragma unroll
  for (int p = 0; p < NP; ++p) {
    atomicAdd(&kwnw[(size_t)p * DDIM + d0], a0[p]);
    atomicAdd(&kwnw[(size_t)p * DDIM + d0 + 1], a1[p]);
  }
}

__global__ void k_div(float* kwnw, const float* __restrict__ colsum) {
  int i = blockIdx.x * 256 + threadIdx.x;
  if (i < NP * DDIM) {
    int p = i >> 9;
    float d = colsum[p];
    kwnw[i] = kwnw[i] / ((d > 0.f) ? d : 1.0f);
  }
}

// ----------------------------------------------------- tiny proto MLPs
__global__ void k_mlp1(const float* __restrict__ kwnw,
                       const float* __restrict__ Wk1, const float* __restrict__ bk1,
                       const float* __restrict__ Wn1, const float* __restrict__ bn1,
                       float* __restrict__ hidden) {
  int p = blockIdx.x;
  int h = blockIdx.y * 128 + threadIdx.x;
  const float* W = (p < NKP) ? Wk1 : Wn1;
  float s = (p < NKP) ? bk1[h] : bn1[h];
  const float* row = kwnw + (size_t)p * DDIM;
  for (int d = 0; d < DDIM; ++d) s += row[d] * W[(size_t)d * HDIM + h];
  hidden[(size_t)p * HDIM + h] = fmaxf(s, 0.f);
}

__global__ void k_mlp2(const float* __restrict__ hidden,
                       const float* __restrict__ Wk2, const float* __restrict__ bk2,
                       const float* __restrict__ Wn2, const float* __restrict__ bn2,
                       const float* __restrict__ kp, const float* __restrict__ nv,
                       float* __restrict__ outK, float* __restrict__ outN) {
  int p = blockIdx.x;
  int d = blockIdx.y * 128 + threadIdx.x;
  const float* W = (p < NKP) ? Wk2 : Wn2;
  float s = (p < NKP) ? bk2[d] : bn2[d];
  float base = (p < NKP) ? kp[(size_t)p * DDIM + d] : nv[(size_t)(p - NKP) * DDIM + d];
  const float* row = hidden + (size_t)p * HDIM;
  for (int h = 0; h < HDIM; ++h) s += row[h] * W[(size_t)h * DDIM + d];
  float o = base + s;
  if (p < NKP) outK[(size_t)p * DDIM + d] = o;
  else         outN[(size_t)(p - NKP) * DDIM + d] = o;
}

__global__ void k_norm30(const float* __restrict__ outK, const float* __restrict__ outN,
                         float* __restrict__ proto2N) {
  int p = blockIdx.x, tid = threadIdx.x;
  const float* src = (p < NKP) ? outK + (size_t)p * DDIM : outN + (size_t)(p - NKP) * DDIM;
  __shared__ float red[256];
  float s = 0.f;
  for (int i = tid; i < DDIM; i += 256) { float v = src[i]; s += v * v; }
  red[tid] = s; __syncthreads();
  for (int o = 128; o > 0; o >>= 1) { if (tid < o) red[tid] += red[tid + o]; __syncthreads(); }
  float inv = 1.0f / fmaxf(sqrtf(red[0]), EPSV);
  for (int i = tid; i < DDIM; i += 256) proto2N[(size_t)p * DDIM + i] = src[i] * inv;
}

// --------------------------------------------- logits + argmax, wave/row
__global__ void k_final(const float* __restrict__ refined,
                        const float* __restrict__ Wkc, const float* __restrict__ bkc,
                        const float* __restrict__ Wnc, const float* __restrict__ bnc,
                        const float* __restrict__ proto2N,
                        float* __restrict__ outKL, float* __restrict__ outNL,
                        float* __restrict__ outLbl) {
  __shared__ float sW[NP * DDIM];
  const int tid = threadIdx.x;
  for (int i = tid; i < NP * DDIM; i += 256) {
    int cc = i >> 9, d = i & 511;
    sW[i] = (cc < NKP) ? Wkc[(size_t)d * NKP + cc] : Wnc[(size_t)d * NVP + (cc - NKP)];
  }
  __syncthreads();
  const int lane = tid & 31, wid = tid >> 5;

  for (int rr = 0; rr < 16; ++rr) {
    const size_t r = (size_t)blockIdx.x * 128 + wid * 16 + rr;
    float x[16];
    const float4* prow = reinterpret_cast<const float4*>(refined + r * DDIM + lane * 16);
#pragma unroll
    for (int q = 0; q < 4; ++q) {
      float4 f = prow[q];
      x[4 * q] = f.x; x[4 * q + 1] = f.y; x[4 * q + 2] = f.z; x[4 * q + 3] = f.w;
    }
    for (int c = 0; c < NP; ++c) {
      const float* wc = &sW[c * DDIM + lane * 16];
      float pa = 0.f;
#pragma unroll
      for (int j = 0; j < 16; ++j) pa += x[j] * wc[j];
      pa = warpSum(pa);
      float val = pa + ((c < NKP) ? bkc[c] : bnc[c - NKP]);
      if (lane == 0) {
        if (c < NKP) outKL[r * NKP + c] = val;
        else         outNL[r * NVP + (c - NKP)] = val;
      }
    }
    float ss = 0.f;
#pragma unroll
    for (int j = 0; j < 16; ++j) ss += x[j] * x[j];
    ss = warpSum(ss);
    const float inv = 1.0f / fmaxf(sqrtf(ss), EPSV);
    float bv = -INFINITY; int bi = 0;
    for (int p = 0; p < NP; ++p) {
      const float* pr = proto2N + (size_t)p * DDIM + lane * 16;
      float pa = 0.f;
#pragma unroll
      for (int j = 0; j < 16; ++j) pa += x[j] * pr[j];
      pa = warpSum(pa) * inv;
      if (pa > bv) { bv = pa; bi = p; }
    }
    if (lane == 0) outLbl[r] = (float)bi;
  }
}

__global__ void k_softmax_pw(const float* __restrict__ colsum, float* __restrict__ out) {
  int l = threadIdx.x;
  float v = (l < NP) ? colsum[l] : -INFINITY;
  float m = v;
#pragma unroll
  for (int o = 16; o > 0; o >>= 1) m = fmaxf(m, __shfl_xor(m, o, 32));
  float e = (l < NP) ? __expf(v - m) : 0.f;
  float s = e;
#pragma unroll
  for (int o = 16; o > 0; o >>= 1) s += __shfl_xor(s, o, 32);
  if (l < NP) out[l] = e / s;
}

// =====================================================================
extern "C" void kernel_launch(void* const* d_in, const int* in_sizes, int n_in,
                              void* d_out, int out_size, void* d_ws, size_t ws_size,
                              hipStream_t stream) {
  (void)in_sizes; (void)n_in; (void)out_size; (void)ws_size;
  const float* pf   = (const float*)d_in[0];
  const float* kp   = (const float*)d_in[1];
  const float* nv   = (const float*)d_in[2];
  const float* W0   = (const float*)d_in[3];
  const float* b0   = (const float*)d_in[4];
  const float* W1   = (const float*)d_in[5];
  const float* b1   = (const float*)d_in[6];
  const float* Wk1  = (const float*)d_in[7];
  const float* bk1  = (const float*)d_in[8];
  const float* Wk2  = (const float*)d_in[9];
  const float* bk2  = (const float*)d_in[10];
  const float* Wn1  = (const float*)d_in[11];
  const float* bn1  = (const float*)d_in[12];
  const float* Wn2  = (const float*)d_in[13];
  const float* bn2  = (const float*)d_in[14];
  const float* Wkc  = (const float*)d_in[15];
  const float* bkc  = (const float*)d_in[16];
  const float* Wnc  = (const float*)d_in[17];
  const float* bnc  = (const float*)d_in[18];
  const float* temp = (const float*)d_in[19];

  float* out  = (float*)d_out;
  float* oRef = out;
  float* oK   = oRef + (size_t)NPTS * DDIM;
  float* oNv  = oK + (size_t)NKP * DDIM;
  float* oKL  = oNv + (size_t)NVP * DDIM;
  float* oNL  = oKL + (size_t)NPTS * NKP;
  float* oLbl = oNL + (size_t)NPTS * NVP;
  float* oAdj = oLbl + (size_t)NPTS;
  float* oPW  = oAdj + (size_t)NPTS * NP;

  char* wsBase = (char*)d_ws;
  size_t off = 0;
  auto carve = [&](size_t bytes) -> void* {
    off = (off + 255) & ~(size_t)255;
    void* p = wsBase + off;
    off += bytes;
    return p;
  };
  float* protoN  = (float*)carve((size_t)NP * DDIM * 4);
  float* colsum  = (float*)carve((size_t)NP * 4);
  float* kwnw    = (float*)carve((size_t)NP * DDIM * 4);
  float* hidden  = (float*)carve((size_t)NP * HDIM * 4);
  float* proto2N = (float*)carve((size_t)NP * DDIM * 4);
  u16* W0t       = (u16*)carve((size_t)DDIM * HDIM * 2);  // N x K bf16
  u16* W1t       = (u16*)carve((size_t)HDIM * DDIM * 2);  // N x K bf16
  u16* a_in      = (u16*)carve((size_t)NPTS * DDIM * 2);
  u16* contrib   = (u16*)carve((size_t)NPTS * DDIM * 2);
  u16* x2        = (u16*)carve((size_t)NPTS * DDIM * 2);

  k_zero<<<1, 64, 0, stream>>>(colsum, NP);
  k_zero<<<(NP * DDIM + 255) / 256, 256, 0, stream>>>(kwnw, NP * DDIM);
  k_norm_protos<<<NP, 256, 0, stream>>>(kp, nv, protoN);
  k_transpose_w<<<dim3(DDIM / 32, HDIM / 32), 256, 0, stream>>>(W0, W0t);
  k_transpose_w<<<dim3(HDIM / 32, DDIM / 32), 256, 0, stream>>>(W1, W1t);

  k_phase1<<<NPTS / 128, 256, 0, stream>>>(pf, kp, nv, protoN, temp, oAdj, colsum, a_in, contrib);

  // GEMM1: x2 = relu((pf+contrib)@W0 + b0) + contrib   (bf16 out)
  k_gemm<true, true, true, false><<<dim3(NPTS / 128, HDIM / 64), 256, 0, stream>>>(
      a_in, W0t, b0, contrib, x2, nullptr);
  // GEMM2: refined = x2 @ W1 + b1                       (f32 out -> d_out)
  k_gemm<false, false, false, true><<<dim3(NPTS / 128, DDIM / 64), 256, 0, stream>>>(
      x2, W1t, b1, nullptr, nullptr, oRef);

  k_proto_proj<<<NPTS / 1024, 256, 0, stream>>>(oAdj, oRef, kwnw);
  k_div<<<(NP * DDIM + 255) / 256, 256, 0, stream>>>(kwnw, colsum);
  k_mlp1<<<dim3(NP, HDIM / 128), 128, 0, stream>>>(kwnw, Wk1, bk1, Wn1, bn1, hidden);
  k_mlp2<<<dim3(NP, DDIM / 128), 128, 0, stream>>>(hidden, Wk2, bk2, Wn2, bn2, kp, nv, oK, oNv);
  k_norm30<<<NP, 256, 0, stream>>>(oK, oNv, proto2N);
  k_final<<<NPTS / 128, 256, 0, stream>>>(oRef, Wkc, bkc, Wnc, bnc, proto2N, oKL, oNL, oLbl);
  k_softmax_pw<<<1, 32, 0, stream>>>(colsum, oPW);
}
